// PointNetFeaturePropagation_73065983639806
// MI455X (gfx1250) — matrix operand order repaired and checked by reading, not compiled, not run on hardware
//
#include <hip/hip_runtime.h>
#include <hip/hip_bf16.h>

// ---------------------------------------------------------------------------
// PointNet Feature Propagation, MI455X (gfx1250) version.
// B=8, C=3, N=8192, S=2048, D=256, IN_CH=512, MLP=[256,256]
// GEMMs run on v_wmma_f32_16x16x32_bf16 (bf16 in, fp32 accumulate).
// Each wave: 1 A (weight) fragment reused over 4 n-subtiles -> 4 independent
// WMMA accumulation chains per K-step (XDL latency hiding, 4x A reuse).
// ---------------------------------------------------------------------------

#define BB   8
#define NN   8192
#define SS   2048
#define DD   256
#define KIN  512
#define OO   256

typedef __attribute__((ext_vector_type(16))) __bf16 v16bf;
typedef __attribute__((ext_vector_type(8)))  __bf16 v8bf;
typedef __attribute__((ext_vector_type(8)))  float  v8f;

// ---------------------------------------------------------------------------
// Kernel 0: zero BN-stat accumulators + convert W0/W1 to bf16.
// stats floats: [0..255]=sum0 [256..511]=sq0 [512..767]=sum1 [768..1023]=sq1
// ---------------------------------------------------------------------------
__global__ __launch_bounds__(256) void prep_kernel(const float* __restrict__ W0f,
                                                   const float* __restrict__ W1f,
                                                   __bf16* __restrict__ w0b,
                                                   __bf16* __restrict__ w1b,
                                                   float* __restrict__ stats) {
  int idx = blockIdx.x * 256 + threadIdx.x;
  if (idx < 1024) {
    stats[idx] = 0.0f;
  } else if (idx < 1024 + OO * KIN) {
    int i = idx - 1024;
    w0b[i] = (__bf16)W0f[i];
  } else {
    int i = idx - 1024 - OO * KIN;
    w1b[i] = (__bf16)W1f[i];
  }
}

// ---------------------------------------------------------------------------
// Kernel 1: 3-NN inverse-distance interpolation + concat -> bf16 [b][n][512]
// xyz2 per batch (3*2048 f32 = 24 KB) staged in LDS (320 KB/WGP available).
// points2 gather target is 16.7 MB total -> L2-resident (192 MB L2).
// ---------------------------------------------------------------------------
__global__ __launch_bounds__(256) void interp_kernel(const float* __restrict__ xyz1,
                                                     const float* __restrict__ xyz2,
                                                     const float* __restrict__ points1,
                                                     const float* __restrict__ points2,
                                                     __bf16* __restrict__ XB) {
  __shared__ float sx[SS];
  __shared__ float sy[SS];
  __shared__ float sz[SS];
  const int b = blockIdx.y;
  const int n = blockIdx.x * 256 + threadIdx.x;

  const float* x2 = xyz2 + (size_t)b * 3 * SS;
  for (int i = threadIdx.x; i < SS; i += 256) {
    sx[i] = x2[i];
    sy[i] = x2[SS + i];
    sz[i] = x2[2 * SS + i];
  }
  __syncthreads();

  const float* x1 = xyz1 + (size_t)b * 3 * NN;
  const float px = x1[n], py = x1[NN + n], pz = x1[2 * NN + n];

  float d0 = 3.4e38f, d1 = 3.4e38f, d2 = 3.4e38f;
  int   i0 = 0, i1 = 0, i2 = 0;
  for (int s = 0; s < SS; ++s) {
    const float dx = px - sx[s];
    const float dy = py - sy[s];
    const float dz = pz - sz[s];
    const float d  = dx * dx + dy * dy + dz * dz;
    if (d < d2) {
      if (d < d1) {
        if (d < d0) { d2 = d1; i2 = i1; d1 = d0; i1 = i0; d0 = d; i0 = s; }
        else        { d2 = d1; i2 = i1; d1 = d;  i1 = s; }
      } else        { d2 = d;  i2 = s; }
    }
  }
  float w0 = 1.0f / (d0 + 1e-8f);
  float w1 = 1.0f / (d1 + 1e-8f);
  float w2 = 1.0f / (d2 + 1e-8f);
  const float wsum = w0 + w1 + w2;
  w0 /= wsum; w1 /= wsum; w2 /= wsum;

  const float* p2  = points2 + (size_t)b * DD * SS;
  const float* p1  = points1 + (size_t)b * DD * NN;
  __bf16*      row = XB + ((size_t)b * NN + n) * KIN;

  for (int d = 0; d < DD; ++d) {
    row[d] = (__bf16)p1[(size_t)d * NN + n];
    const float v = w0 * p2[(size_t)d * SS + i0] +
                    w1 * p2[(size_t)d * SS + i1] +
                    w2 * p2[(size_t)d * SS + i2];
    row[DD + d] = (__bf16)v;
  }
}

// ---------------------------------------------------------------------------
// Kernel 2: GEMM via v_wmma_f32_16x16x32_bf16.
//   Y[b][o][n] = sum_k W[o][k] * X[b][n][k] + bias[o]
// W: [OO][K] bf16 row-major (k contiguous); X: [b][NN][K] bf16 (k contiguous).
// Block = 8 waves; each wave owns a 16(o) x 64(n) strip = 4 accumulators.
// Fragment layouts used here (wave32, same contiguous k order in A and B):
//   A 16x32 : lanes 0-15 k 0..15, lanes 16-31 k 16..31 (contiguous), o=lane%16
//   B 32x16 : lanes 0-15 k 0..15, lanes 16-31 k 16..31 (contiguous), n=lane%16
//   C/D     : elem r -> o = otile + r + 8*(lane>=16), n = ntile + lane%16
// ---------------------------------------------------------------------------
template <int K>
__global__ __launch_bounds__(256) void gemm_wmma_kernel(const __bf16* __restrict__ Wb,
                                                        const __bf16* __restrict__ Xb,
                                                        const float* __restrict__ bias,
                                                        float* __restrict__ Y) {
  const int lane   = threadIdx.x & 31;
  const int wave   = threadIdx.x >> 5;
  const int b      = blockIdx.z;
  const int otile  = blockIdx.y * 16;
  const int ntile0 = (blockIdx.x * 8 + wave) * 64;   // 4 x 16-n subtiles
  const int lmod   = lane & 15;
  const int hi     = lane >> 4;                      // 0: lanes 0-15, 1: lanes 16-31

  const int abase = hi ? 16 : 0;
  const int bbase = hi ? 16 : 0;

  const __bf16* wrow = Wb + (size_t)(otile + lmod) * K;
  const __bf16* xrow0 = Xb + ((size_t)b * NN + (ntile0 + lmod)) * K;
  const __bf16* xrow1 = xrow0 + (size_t)16 * K;
  const __bf16* xrow2 = xrow0 + (size_t)32 * K;
  const __bf16* xrow3 = xrow0 + (size_t)48 * K;

  v8f acc0 = {}, acc1 = {}, acc2 = {}, acc3 = {};

#pragma unroll 2
  for (int kb = 0; kb < K; kb += 32) {
    // Speculative prefetch (translation failures silently dropped) -> no branch.
    __builtin_prefetch(xrow0 + kb + 128, 0, 1);      // global_prefetch_b8
    __builtin_prefetch(xrow2 + kb + 128, 0, 1);

    const v8bf alo = *(const v8bf*)(wrow + kb + abase);
    const v8bf ahi = *(const v8bf*)(wrow + kb + abase + 8);
    v16bf A;
#pragma unroll
    for (int i = 0; i < 8; ++i) { A[i] = alo[i]; A[8 + i] = ahi[i]; }

    const v16bf B0 = *(const v16bf*)(xrow0 + kb + bbase);
    const v16bf B1 = *(const v16bf*)(xrow1 + kb + bbase);
    const v16bf B2 = *(const v16bf*)(xrow2 + kb + bbase);
    const v16bf B3 = *(const v16bf*)(xrow3 + kb + bbase);

    acc0 = __builtin_amdgcn_wmma_f32_16x16x32_bf16(false, A, false, B0, (short)0, acc0, false, false);
    acc1 = __builtin_amdgcn_wmma_f32_16x16x32_bf16(false, A, false, B1, (short)0, acc1, false, false);
    acc2 = __builtin_amdgcn_wmma_f32_16x16x32_bf16(false, A, false, B2, (short)0, acc2, false, false);
    acc3 = __builtin_amdgcn_wmma_f32_16x16x32_bf16(false, A, false, B3, (short)0, acc3, false, false);
  }

  const size_t outb = (size_t)b * OO * NN;
#pragma unroll
  for (int r = 0; r < 8; ++r) {
    const int o = otile + r + hi * 8;
    const float bv = bias[o];
    float* yrow = Y + outb + (size_t)o * NN + ntile0 + lmod;
    yrow[0]  = acc0[r] + bv;
    yrow[16] = acc1[r] + bv;
    yrow[32] = acc2[r] + bv;
    yrow[48] = acc3[r] + bv;
  }
}

// ---------------------------------------------------------------------------
// Kernel 3: per-channel sum / sum-of-squares over [b][o][:] (atomic combine).
// ---------------------------------------------------------------------------
__global__ __launch_bounds__(256) void stats_kernel(const float* __restrict__ Y,
                                                    float* __restrict__ sum,
                                                    float* __restrict__ sq) {
  __shared__ float s1[256];
  __shared__ float s2[256];
  const int o = blockIdx.x, b = blockIdx.y, tid = threadIdx.x;
  const float* p = Y + ((size_t)b * OO + o) * NN;
  float a = 0.0f, q = 0.0f;
  for (int n = tid; n < NN; n += 256) {
    const float v = p[n];
    a += v; q += v * v;
  }
  s1[tid] = a; s2[tid] = q;
  __syncthreads();
  for (int st = 128; st > 0; st >>= 1) {
    if (tid < st) { s1[tid] += s1[tid + st]; s2[tid] += s2[tid + st]; }
    __syncthreads();
  }
  if (tid == 0) {
    atomicAdd(&sum[o], s1[0]);
    atomicAdd(&sq[o],  s2[0]);
  }
}

// ---------------------------------------------------------------------------
// Kernel 4: BN(train) + ReLU, transpose to bf16 [b][n][o] for the next GEMM.
// ---------------------------------------------------------------------------
__global__ __launch_bounds__(256) void bn_apply_tr_kernel(const float* __restrict__ Y,
                                                          const float* __restrict__ sum,
                                                          const float* __restrict__ sq,
                                                          const float* __restrict__ g,
                                                          const float* __restrict__ be,
                                                          __bf16* __restrict__ X1) {
  const int idx  = blockIdx.x * 256 + threadIdx.x;
  const int o    = idx & 255;
  const int rest = idx >> 8;
  const int n    = rest & (NN - 1);
  const int b    = rest >> 13;
  const float invBN = 1.0f / (float)(BB * NN);
  const float m = sum[o] * invBN;
  const float v = sq[o] * invBN - m * m;
  const float a = g[o] * rsqrtf(v + 1e-5f);
  const float c = be[o] - a * m;
  const float y = Y[((size_t)b * OO + o) * NN + n];
  const float r = fmaxf(a * y + c, 0.0f);
  X1[((size_t)b * NN + n) * OO + o] = (__bf16)r;
}

// ---------------------------------------------------------------------------
// Kernel 5: final BN(train) + ReLU in place on fp32 [b][o][n] output.
// ---------------------------------------------------------------------------
__global__ __launch_bounds__(256) void bn_apply_inplace_kernel(float* __restrict__ Y,
                                                               const float* __restrict__ sum,
                                                               const float* __restrict__ sq,
                                                               const float* __restrict__ g,
                                                               const float* __restrict__ be) {
  const size_t idx = (size_t)blockIdx.x * 256 + threadIdx.x;
  const int o = (int)((idx >> 13) & 255);
  const float invBN = 1.0f / (float)(BB * NN);
  const float m = sum[o] * invBN;
  const float v = sq[o] * invBN - m * m;
  const float a = g[o] * rsqrtf(v + 1e-5f);
  const float c = be[o] - a * m;
  Y[idx] = fmaxf(a * Y[idx] + c, 0.0f);
}

// ---------------------------------------------------------------------------
extern "C" void kernel_launch(void* const* d_in, const int* in_sizes, int n_in,
                              void* d_out, int out_size, void* d_ws, size_t ws_size,
                              hipStream_t stream) {
  (void)in_sizes; (void)n_in; (void)out_size; (void)ws_size;

  const float* xyz1    = (const float*)d_in[0];
  const float* xyz2    = (const float*)d_in[1];
  const float* points1 = (const float*)d_in[2];
  const float* points2 = (const float*)d_in[3];
  const float* W0f     = (const float*)d_in[4];
  const float* b0      = (const float*)d_in[5];
  const float* W1f     = (const float*)d_in[6];
  const float* b1      = (const float*)d_in[7];
  const float* g0      = (const float*)d_in[8];
  const float* be0     = (const float*)d_in[9];
  const float* g1      = (const float*)d_in[10];
  const float* be1     = (const float*)d_in[11];

  float* out = (float*)d_out;   // [8][256][8192] fp32; doubles as GEMM-0 scratch

  // Workspace layout (bytes, 256-aligned), total ~101 MB:
  //   w0b   : 256*512*2          = 262144
  //   w1b   : 256*256*2          = 131072
  //   stats : 1024 f32           = 4096
  //   xb    : 8*8192*512 bf16    = 67108864   [b][n][512]
  //   x1b   : 8*8192*256 bf16    = 33554432   [b][n][256]
  char* ws = (char*)d_ws;
  __bf16* w0b   = (__bf16*)(ws);
  __bf16* w1b   = (__bf16*)(ws + 262144);
  float*  stats = (float*)(ws + 262144 + 131072);
  __bf16* xb    = (__bf16*)(ws + 262144 + 131072 + 4096);
  __bf16* x1b   = (__bf16*)(ws + 262144 + 131072 + 4096 + (size_t)67108864);

  float* sum0 = stats;
  float* sq0  = stats + 256;
  float* sum1 = stats + 512;
  float* sq1  = stats + 768;

  // 0) zero stats, convert weights to bf16   (1024 + 131072 + 65536 = 197632)
  prep_kernel<<<197632 / 256, 256, 0, stream>>>(W0f, W1f, w0b, w1b, stats);

  // 1) 3-NN interpolation + concat -> xb [b][n][512] bf16
  interp_kernel<<<dim3(NN / 256, BB), 256, 0, stream>>>(xyz1, xyz2, points1, points2, xb);

  // 2) GEMM-0 (K=512) via bf16 WMMA -> out (scratch, fp32 pre-BN)
  gemm_wmma_kernel<KIN><<<dim3(NN / 512, OO / 16, BB), 256, 0, stream>>>(w0b, xb, b0, out);

  // 3) BN-0 stats
  stats_kernel<<<dim3(OO, BB), 256, 0, stream>>>(out, sum0, sq0);

  // 4) BN-0 + ReLU, transpose -> x1b [b][n][256] bf16
  bn_apply_tr_kernel<<<(BB * OO * NN) / 256, 256, 0, stream>>>(out, sum0, sq0, g0, be0, x1b);

  // 5) GEMM-1 (K=256) via bf16 WMMA -> out (fp32 pre-BN)
  gemm_wmma_kernel<DD><<<dim3(NN / 512, OO / 16, BB), 256, 0, stream>>>(w1b, x1b, b1, out);

  // 6) BN-1 stats
  stats_kernel<<<dim3(OO, BB), 256, 0, stream>>>(out, sum1, sq1);

  // 7) BN-1 + ReLU in place -> final output
  bn_apply_inplace_kernel<<<(BB * OO * NN) / 256, 256, 0, stream>>>(out, sum1, sq1, g1, be1);
}
